// SparseCausalAttentionProcessor_10514079941017
// MI455X (gfx1250) — compile-verified, matrix operand order checked by reading, not run on hardware
//
#include <hip/hip_runtime.h>

// ---------------------------------------------------------------------------
// SparseCausalAttention for MI455X (gfx1250), bf16 WMMA pipeline.
//   q    = hidden @ Wq + bq                    [32768, 640]
//   k/v  = hidden[frame0 rows] @ Wk/Wv + b     [ 2048, 640]
//   attn = flash-attention per (b,f,h)         [32768, 640]  (c = h*80+d)
//   out  = attn @ Wo + bo + hidden             [32768, 640]
// Workspace usage: ~178 MB in d_ws (q, k, v, attn), fully written before read.
// Double-buffered LDS staging in both kernels: global loads for tile t+1 are
// issued before the WMMA work on tile t (one barrier per iteration).
// ---------------------------------------------------------------------------

typedef __attribute__((ext_vector_type(16))) __bf16 v16bf;
typedef __attribute__((ext_vector_type(8)))  float  v8f;

union Frag16 {
    uint4 q[2];   // two 16-byte LDS loads
    v16bf v;      // 16 bf16 lanes for WMMA
};

__device__ __forceinline__ unsigned short f2bf(float x) {
    unsigned int u = __float_as_uint(x);
    u += 0x7FFFu + ((u >> 16) & 1u);        // round-to-nearest-even
    return (unsigned short)(u >> 16);
}

// ---------------------------------------------------------------------------
// Tiled GEMM: Out[M,N] = A'[M,K] @ W[K,N] + bias[N] (+ resid[M,N])
// A' row m reads A row ((m/rowGroup)*rowStride + m%rowGroup)  (frame-0 gather)
// BM=BN=128, BK=32, 256 threads = 8 waves (4x2), wave tile 32x64 (2x4 frags).
// ---------------------------------------------------------------------------
#define GSTR 48   // LDS row stride in bf16 elems (96B, 16B-aligned, pads banks)

template <bool RESID>
__global__ __launch_bounds__(256)
void gemm_bias_kernel(const float* __restrict__ A, const float* __restrict__ W,
                      const float* __restrict__ bias, const float* __restrict__ resid,
                      float* __restrict__ Out,
                      int M, int N, int K, int rowGroup, int rowStride)
{
    __shared__ unsigned short As[2][128 * GSTR];
    __shared__ unsigned short Bs[2][128 * GSTR];

    const int tid  = threadIdx.x;
    const int wave = tid >> 5, lane = tid & 31;
    const int wm = wave & 3, wn = wave >> 2;           // 4 x 2 wave grid
    const int lr = lane & 15, lh = lane >> 4;
    const int bm = blockIdx.y, bn = blockIdx.x;

    // Per-thread staging coordinates (loop-invariant, incl. frame-0 row map).
    int    a_r[4], a_c[4], b_k[4], b_n[4];
    size_t a_row[4];
    #pragma unroll
    for (int i = 0; i < 4; ++i) {
        int idx = tid + i * 256;                // 0..1023
        a_r[i] = idx >> 3;                      // A row 0..127
        a_c[i] = (idx & 7) << 2;                // A col 0,4,..,28
        int gm = bm * 128 + a_r[i];
        a_row[i] = (size_t)(gm / rowGroup) * rowStride + (gm % rowGroup);
        b_k[i] = idx >> 5;                      // W k 0..31
        b_n[i] = (idx & 31) << 2;               // W n 0..124 step 4
    }

    auto stage = [&](int k0, int buf) {
        #pragma unroll
        for (int i = 0; i < 4; ++i) {
            float4 f = *(const float4*)(A + a_row[i] * K + k0 + a_c[i]);
            unsigned short* dst = &As[buf][a_r[i] * GSTR + a_c[i]];
            dst[0] = f2bf(f.x); dst[1] = f2bf(f.y);
            dst[2] = f2bf(f.z); dst[3] = f2bf(f.w);
        }
        #pragma unroll
        for (int i = 0; i < 4; ++i) {
            float4 f = *(const float4*)(W + (size_t)(k0 + b_k[i]) * N + bn * 128 + b_n[i]);
            unsigned short* dst = &Bs[buf][b_n[i] * GSTR + b_k[i]];
            dst[0 * GSTR] = f2bf(f.x);
            dst[1 * GSTR] = f2bf(f.y);
            dst[2 * GSTR] = f2bf(f.z);
            dst[3 * GSTR] = f2bf(f.w);
        }
    };

    const int nk = K / 32;
    v8f acc[2][4] = {};

    stage(0, 0);
    __syncthreads();

    for (int kt = 0; kt < nk; ++kt) {
        if (kt + 1 < nk) stage((kt + 1) * 32, (kt + 1) & 1);   // prefetch next

        const unsigned short* Asb = As[kt & 1];
        const unsigned short* Bsb = Bs[kt & 1];

        // Fragments (ISA 16-bit A/B layout: lane half picks K sub-run).
        Frag16 a[2], b[4];
        const int kb = lh * 8;
        #pragma unroll
        for (int mi = 0; mi < 2; ++mi) {
            int r = wm * 32 + mi * 16 + lr;
            a[mi].q[0] = *(const uint4*)&Asb[r * GSTR + kb];
            a[mi].q[1] = *(const uint4*)&Asb[r * GSTR + kb + 16];
        }
        #pragma unroll
        for (int ni = 0; ni < 4; ++ni) {
            int c = wn * 64 + ni * 16 + lr;
            b[ni].q[0] = *(const uint4*)&Bsb[c * GSTR + kb];
            b[ni].q[1] = *(const uint4*)&Bsb[c * GSTR + kb + 16];
        }
        #pragma unroll
        for (int mi = 0; mi < 2; ++mi)
            #pragma unroll
            for (int ni = 0; ni < 4; ++ni)
                acc[mi][ni] = __builtin_amdgcn_wmma_f32_16x16x32_bf16(
                    false, a[mi].v, false, b[ni].v, (short)0, acc[mi][ni],
                    false, false);
        __syncthreads();
    }

    // Epilogue: f32 C layout (VGPR r -> row r + 8*lh, lane&15 -> col).
    #pragma unroll
    for (int mi = 0; mi < 2; ++mi)
        #pragma unroll
        for (int ni = 0; ni < 4; ++ni) {
            int gn = bn * 128 + wn * 64 + ni * 16 + lr;
            float bv = bias[gn];
            #pragma unroll
            for (int r = 0; r < 8; ++r) {
                int m = bm * 128 + wm * 32 + mi * 16 + r + lh * 8;
                float val = acc[mi][ni][r] + bv;
                if constexpr (RESID) val += resid[(size_t)m * N + gn];
                Out[(size_t)m * N + gn] = val;
            }
        }
}

// ---------------------------------------------------------------------------
// Flash attention: one block = (batch-row Br, head h, 128-query tile qt).
// D=80 padded to 96 for the QK^T k-loop (3 x 32). 8 waves, 16 q-rows each.
// K/V tiles double-buffered.
// ---------------------------------------------------------------------------
#define SQ 1024
#define DH 80
#define DP 96
#define TT 128

__global__ __launch_bounds__(256)
void flash_attn_kernel(const float* __restrict__ Q, const float* __restrict__ Kb,
                       const float* __restrict__ Vb, float* __restrict__ O)
{
    __shared__ unsigned short Qs[TT * DP];          // 24 KB, q rows x padded dims
    __shared__ unsigned short Ks[2][TT * DP];       // 48 KB, keys  x padded dims
    __shared__ unsigned short Vs[2][DH * TT];       // 40 KB, dims  x keys (B layout)
    __shared__ unsigned short Ps[8 * 16 * TT];      // 32 KB, per-wave P scratch

    const int qt = blockIdx.x;        // 0..7   query tile
    const int h  = blockIdx.y;        // 0..7   head
    const int Br = blockIdx.z;        // 0..31  flat batch row (video*16+frame)
    const int bi = Br >> 4;           // video index (f = 16)

    const int tid = threadIdx.x, wave = tid >> 5, lane = tid & 31;
    const int lr = lane & 15, lh = lane >> 4;

    const size_t qbase  = ((size_t)Br * SQ + (size_t)qt * TT) * 640 + h * DH;
    const size_t kvbase = (size_t)bi * SQ * 640 + h * DH;
    const float  scale  = 0.111803398874989485f;   // 1/sqrt(80)

    // Loop-invariant staging coordinates.
    int kr[12], kdq[12], vt[10], vdq[10];
    #pragma unroll
    for (int i = 0; i < 12; ++i) {
        int idx = tid + i * 256;                    // 128 rows * 24 quads
        kr[i] = idx / 24; kdq[i] = (idx % 24) * 4;
    }
    #pragma unroll
    for (int i = 0; i < 10; ++i) {
        int idx = tid + i * 256;                    // 128 keys * 20 quads
        vt[i] = idx / 20; vdq[i] = (idx % 20) * 4;
    }

    auto stage_kv = [&](int j, int buf) {
        #pragma unroll
        for (int i = 0; i < 12; ++i) {              // K tile: keys x padded dims
            unsigned short* dst = &Ks[buf][kr[i] * DP + kdq[i]];
            if (kdq[i] < DH) {
                float4 f = *(const float4*)(Kb + kvbase + (size_t)(j * TT + kr[i]) * 640 + kdq[i]);
                dst[0] = f2bf(f.x); dst[1] = f2bf(f.y);
                dst[2] = f2bf(f.z); dst[3] = f2bf(f.w);
            } else {
                dst[0] = 0; dst[1] = 0; dst[2] = 0; dst[3] = 0;
            }
        }
        #pragma unroll
        for (int i = 0; i < 10; ++i) {              // V tile transposed [d][key]
            float4 f = *(const float4*)(Vb + kvbase + (size_t)(j * TT + vt[i]) * 640 + vdq[i]);
            unsigned short* dst = &Vs[buf][vdq[i] * TT + vt[i]];
            dst[0 * TT] = f2bf(f.x);
            dst[1 * TT] = f2bf(f.y);
            dst[2 * TT] = f2bf(f.z);
            dst[3 * TT] = f2bf(f.w);
        }
    };

    // --- stage Q tile (zero-pad dims 80..95) and first K/V tile ------------
    #pragma unroll
    for (int i = 0; i < 12; ++i) {
        unsigned short* dst = &Qs[kr[i] * DP + kdq[i]];
        if (kdq[i] < DH) {
            float4 f = *(const float4*)(Q + qbase + (size_t)kr[i] * 640 + kdq[i]);
            dst[0] = f2bf(f.x); dst[1] = f2bf(f.y);
            dst[2] = f2bf(f.z); dst[3] = f2bf(f.w);
        } else {
            dst[0] = 0; dst[1] = 0; dst[2] = 0; dst[3] = 0;
        }
    }
    stage_kv(0, 0);
    __syncthreads();

    // Q A-fragments are invariant over the whole key loop.
    Frag16 qa[3];
    #pragma unroll
    for (int ks = 0; ks < 3; ++ks) {
        int r = wave * 16 + lr;
        int kk = ks * 32 + lh * 8;
        qa[ks].q[0] = *(const uint4*)&Qs[r * DP + kk];
        qa[ks].q[1] = *(const uint4*)&Qs[r * DP + kk + 16];
    }

    float mrow[8], lrow[8];
    v8f acc[5] = {};
    #pragma unroll
    for (int r = 0; r < 8; ++r) { mrow[r] = -1e30f; lrow[r] = 0.f; }

    for (int j = 0; j < SQ / TT; ++j) {
        if (j + 1 < SQ / TT) stage_kv(j + 1, (j + 1) & 1);     // prefetch next

        const unsigned short* Ksb = Ks[j & 1];
        const unsigned short* Vsb = Vs[j & 1];

        // --- S = Q @ K^T (16 q-rows x 128 keys per wave) -------------------
        v8f s[8] = {};
        #pragma unroll
        for (int ks = 0; ks < 3; ++ks)
            #pragma unroll
            for (int nt = 0; nt < 8; ++nt) {
                Frag16 kf;
                int c = nt * 16 + lr;
                int kk = ks * 32 + lh * 8;
                kf.q[0] = *(const uint4*)&Ksb[c * DP + kk];
                kf.q[1] = *(const uint4*)&Ksb[c * DP + kk + 16];
                s[nt] = __builtin_amdgcn_wmma_f32_16x16x32_bf16(
                    false, qa[ks].v, false, kf.v, (short)0, s[nt], false, false);
            }

        // --- online softmax, per row r (row = r + 8*lh within wave tile) ---
        #pragma unroll
        for (int r = 0; r < 8; ++r) {
            float mx = -1e30f;
            #pragma unroll
            for (int nt = 0; nt < 8; ++nt) mx = fmaxf(mx, s[nt][r]);
            #pragma unroll
            for (int msk = 1; msk < 16; msk <<= 1)
                mx = fmaxf(mx, __shfl_xor(mx, msk, 32));
            float mnew = fmaxf(mrow[r], mx * scale);

            float sum = 0.f;
            #pragma unroll
            for (int nt = 0; nt < 8; ++nt) {
                float p = __expf(s[nt][r] * scale - mnew);
                sum += p;
                s[nt][r] = p;
            }
            #pragma unroll
            for (int msk = 1; msk < 16; msk <<= 1)
                sum += __shfl_xor(sum, msk, 32);

            float alpha = __expf(mrow[r] - mnew);
            lrow[r] = lrow[r] * alpha + sum;
            mrow[r] = mnew;
            #pragma unroll
            for (int d = 0; d < 5; ++d) acc[d][r] *= alpha;
        }

        // --- spill P (C layout) to wave-private LDS as bf16 ----------------
        #pragma unroll
        for (int nt = 0; nt < 8; ++nt)
            #pragma unroll
            for (int r = 0; r < 8; ++r)
                Ps[(wave * 16 + r + lh * 8) * TT + nt * 16 + lr] = f2bf(s[nt][r]);

        // --- acc += P @ V (same-wave LDS RAW, DS ops are in-order) ---------
        #pragma unroll
        for (int ks = 0; ks < 4; ++ks) {
            Frag16 pa;
            int m = wave * 16 + lr;
            int kk = ks * 32 + lh * 8;
            pa.q[0] = *(const uint4*)&Ps[m * TT + kk];
            pa.q[1] = *(const uint4*)&Ps[m * TT + kk + 16];
            #pragma unroll
            for (int nt = 0; nt < 5; ++nt) {
                Frag16 vf;
                int d = nt * 16 + lr;
                vf.q[0] = *(const uint4*)&Vsb[d * TT + kk];
                vf.q[1] = *(const uint4*)&Vsb[d * TT + kk + 16];
                acc[nt] = __builtin_amdgcn_wmma_f32_16x16x32_bf16(
                    false, pa.v, false, vf.v, (short)0, acc[nt], false, false);
            }
        }
        __syncthreads();
    }

    // --- normalize and write out: O[row, h*80 + d] -------------------------
    #pragma unroll
    for (int nt = 0; nt < 5; ++nt)
        #pragma unroll
        for (int r = 0; r < 8; ++r) {
            int m = r + lh * 8;
            size_t row = (size_t)Br * SQ + (size_t)qt * TT + wave * 16 + m;
            O[row * 640 + h * DH + nt * 16 + lr] = acc[nt][r] / lrow[r];
        }
}

// ---------------------------------------------------------------------------
extern "C" void kernel_launch(void* const* d_in, const int* in_sizes, int n_in,
                              void* d_out, int out_size, void* d_ws, size_t ws_size,
                              hipStream_t stream)
{
    (void)in_sizes; (void)n_in; (void)out_size; (void)ws_size;
    const float* hidden = (const float*)d_in[0];
    const float* Wq = (const float*)d_in[1];
    const float* Wk = (const float*)d_in[2];
    const float* Wv = (const float*)d_in[3];
    const float* Wo = (const float*)d_in[4];
    const float* bq = (const float*)d_in[5];
    const float* bk = (const float*)d_in[6];
    const float* bv = (const float*)d_in[7];
    const float* bo = (const float*)d_in[8];

    const int B = 32, S = 1024, C = 640, F = 16;
    const int MQ = B * S;          // 32768
    const int MKV = (B / F) * S;   // 2048

    float* q    = (float*)d_ws;                 // 32768*640
    float* k    = q + (size_t)MQ * C;           //  2048*640
    float* v    = k + (size_t)MKV * C;          //  2048*640
    float* attn = v + (size_t)MKV * C;          // 32768*640

    dim3 blk(256);

    // Q projection: all rows, identity row map.
    gemm_bias_kernel<false><<<dim3(C / 128, MQ / 128), blk, 0, stream>>>(
        hidden, Wq, bq, nullptr, q, MQ, C, C, MQ, MQ);
    // K/V projections: frame-0 rows only (row m -> (m/1024)*16384 + m%1024).
    gemm_bias_kernel<false><<<dim3(C / 128, MKV / 128), blk, 0, stream>>>(
        hidden, Wk, bk, nullptr, k, MKV, C, C, S, F * S);
    gemm_bias_kernel<false><<<dim3(C / 128, MKV / 128), blk, 0, stream>>>(
        hidden, Wv, bv, nullptr, v, MKV, C, C, S, F * S);

    // Attention: grid = (q-tiles, heads, flat batch rows).
    flash_attn_kernel<<<dim3(S / 128, 8, B), blk, 0, stream>>>(q, k, v, attn);

    // Output projection + bias + residual.
    gemm_bias_kernel<true><<<dim3(C / 128, MQ / 128), blk, 0, stream>>>(
        attn, Wo, bo, hidden, (float*)d_out, MQ, C, C, MQ, MQ);
}
